// StyleTextToSpeech2AdainResBlock1d_78108275245164
// MI455X (gfx1250) — matrix operand order repaired and checked by reading, not compiled, not run on hardware
//
#include <hip/hip_runtime.h>
#include <hip/hip_bf16.h>
#include <math.h>

typedef __attribute__((ext_vector_type(16))) _Float16 v16h;
typedef __attribute__((ext_vector_type(8)))  _Float16 v8h;
typedef __attribute__((ext_vector_type(8)))  float    v8f;

#define B_   8
#define S_   2048
#define SP_  2050          // S + 2 (zero pad row on each side)
#define H_   512
#define STY_ 128
#define LEAKY_ 0.2f
#define EPS_ 1e-5f
#define RSQRT2_ 0.70710678118654752440f

// ---------------------------------------------------------------------------
// 1) weight-norm scale: scale[c][o] = g[o] / sqrt(sum_{i,k} v[o][i][k]^2)
// ---------------------------------------------------------------------------
__global__ __launch_bounds__(256)
void wn_scale_kernel(const float* __restrict__ v1, const float* __restrict__ g1,
                     const float* __restrict__ v2, const float* __restrict__ g2,
                     float* __restrict__ scale)
{
    int idx = blockIdx.x * 256 + threadIdx.x;        // 1024 total
    if (idx >= 2 * H_) return;
    int c = idx >> 9, o = idx & 511;
    const float* v = c ? v2 : v1;
    const float* g = c ? g2 : g1;
    const float* row = v + (size_t)o * (H_ * 3);
    float ss = 0.f;
    for (int i = 0; i < H_ * 3; ++i) { float t = row[i]; ss += t * t; }
    scale[idx] = g[o] / sqrtf(ss);
}

// ---------------------------------------------------------------------------
// 2) repack normalized weights to f16, tap-major: Wh[c][t][o][i]
// ---------------------------------------------------------------------------
__global__ __launch_bounds__(256)
void wn_write_kernel(const float* __restrict__ v1, const float* __restrict__ v2,
                     const float* __restrict__ scale, _Float16* __restrict__ wh)
{
    size_t idx = (size_t)blockIdx.x * 256 + threadIdx.x; // 2*3*512*512
    int i  = (int)(idx & 511);
    int o  = (int)((idx >> 9) & 511);
    int ct = (int)(idx >> 18);           // 0..5 = c*3+t
    int t  = ct % 3, c = ct / 3;
    const float* v = c ? v2 : v1;
    float w = scale[c * H_ + o] * v[(size_t)o * (H_ * 3) + (size_t)i * 3 + t];
    wh[idx] = (_Float16)w;
}

// ---------------------------------------------------------------------------
// 3) style projections: hs[p][b][j] = style[b] . projp_w[j] + projp_b[j]
// ---------------------------------------------------------------------------
__global__ __launch_bounds__(256)
void proj_kernel(const float* __restrict__ style,
                 const float* __restrict__ w1, const float* __restrict__ b1,
                 const float* __restrict__ w2, const float* __restrict__ b2,
                 float* __restrict__ hs)
{
    int idx = blockIdx.x * 256 + threadIdx.x;        // 2*8*1024 = 16384
    int p = idx >> 13, b = (idx >> 10) & 7, j = idx & 1023;
    const float* w  = p ? w2 : w1;
    const float* bb = p ? b2 : b1;
    const float* st = style + b * STY_;
    const float* wr = w + (size_t)j * STY_;
    float acc = bb[j];
    for (int k = 0; k < STY_; ++k) acc += st[k] * wr[k];
    hs[idx] = acc;
}

// ---------------------------------------------------------------------------
// 4) masked per-(b,h) instance-norm stats
// ---------------------------------------------------------------------------
__global__ __launch_bounds__(256)
void stats_kernel(const float* __restrict__ x, const int* __restrict__ lengths,
                  float* __restrict__ mean, float* __restrict__ rstd)
{
    __shared__ float ssum[256], ssq[256];
    int b  = blockIdx.x >> 3;            // 8 h-groups of 64
    int hg = blockIdx.x & 7;
    int hh = threadIdx.x & 63;
    int sg = threadIdx.x >> 6;           // 0..3
    int h  = hg * 64 + hh;
    int len = lengths[b];
    const float* p = x + (size_t)b * S_ * H_ + h;
    float s = 0.f, q = 0.f;
    for (int si = sg; si < len; si += 4) {
        float v = p[(size_t)si * H_];
        s += v; q += v * v;
    }
    ssum[threadIdx.x] = s; ssq[threadIdx.x] = q;
    __syncthreads();
    if (sg == 0) {
        for (int g = 1; g < 4; ++g) { s += ssum[hh + 64 * g]; q += ssq[hh + 64 * g]; }
        float inv = 1.f / (float)len;
        float m = s * inv;
        float var = q * inv - m * m;
        mean[b * H_ + h] = m;
        rstd[b * H_ + h] = rsqrtf(var + EPS_);
    }
}

// ---------------------------------------------------------------------------
// 5) normalize + AdaIN affine + mask + leaky -> padded f16 [B][S+2][H]
// ---------------------------------------------------------------------------
__global__ __launch_bounds__(256)
void norm_kernel(const float* __restrict__ x, const float* __restrict__ hs,
                 const float* __restrict__ mean, const float* __restrict__ rstd,
                 const int* __restrict__ lengths, _Float16* __restrict__ xh)
{
    size_t idx = (size_t)blockIdx.x * 256 + threadIdx.x; // B*(S+2)*H
    int h   = (int)(idx & 511);
    int row = (int)((idx >> 9) % SP_);
    int b   = (int)(idx / ((size_t)SP_ * H_));
    int s   = row - 1;
    float o = 0.f;
    if (s >= 0 && s < S_ && s < lengths[b]) {
        float v  = x[((size_t)b * S_ + s) * H_ + h];
        float xn = (v - mean[b * H_ + h]) * rstd[b * H_ + h];
        float g  = hs[b * (2 * H_) + h];
        float be = hs[b * (2 * H_) + H_ + h];
        o = (1.f + g) * xn + be;
        o = (o >= 0.f) ? o : LEAKY_ * o;
    }
    xh[idx] = (_Float16)o;
}

// ---------------------------------------------------------------------------
// 6) K=3 conv1d as WMMA f16 GEMM with 3 row-shifted taps.
//    Wave computes 16(M) x 64(N); block = 8 waves = 32(M) x 256(N).
//    A frag: lanes 0-15 -> K 0..15 ; lanes 16-31 -> K 16..31
//    B frag: Wh[t][n][k], 16 contiguous K per lane (lane&15 = N column)
// ---------------------------------------------------------------------------
__global__ __launch_bounds__(256)
void gemm_conv3_kernel(const _Float16* __restrict__ xh,  // [B][S+2][H] f16
                       const _Float16* __restrict__ wh,  // [3][H][H] f16
                       const float* __restrict__ bias,   // [H]
                       const float* __restrict__ hidden, // residual (mode 1)
                       const int* __restrict__ lengths,
                       float* __restrict__ out,          // [B][S][H] f32
                       int mode)
{
    const int lane = threadIdx.x & 31;
    const int w    = threadIdx.x >> 5;       // 0..7
    const int wm   = w & 1;
    const int wn   = w >> 1;                 // 0..3
    const int mt   = blockIdx.x * 2 + wm;    // 16-row tile id, 0..1023
    const int b    = mt >> 7;                // 128 tiles per batch
    const int s0   = (mt & 127) << 4;
    const int nb   = blockIdx.y * 256 + wn * 64;

    const int Ml = lane & 15;
    const int hi = lane >> 4;

    const _Float16* aBase = xh + ((size_t)b * SP_ + s0 + Ml) * H_ + hi * 16;

    v8f acc[4] = {};

    #pragma unroll 1
    for (int kb = 0; kb < H_ / 32; ++kb) {
        const int K0 = kb * 32;
        v16h a[3];
        #pragma unroll
        for (int t = 0; t < 3; ++t) {
            const _Float16* p = aBase + (size_t)t * H_ + K0;
            v8h lo = *(const v8h*)p;
            v8h hh = *(const v8h*)(p + 8);
            a[t] = __builtin_shufflevector(lo, hh, 0,1,2,3,4,5,6,7,
                                                   8,9,10,11,12,13,14,15);
        }
        #pragma unroll
        for (int nt = 0; nt < 4; ++nt) {
            const int n = nb + nt * 16 + Ml;
            #pragma unroll
            for (int t = 0; t < 3; ++t) {
                const _Float16* pb = wh + ((size_t)t * H_ + n) * H_ + K0 + hi * 16;
                v16h bf = *(const v16h*)pb;
                acc[nt] = __builtin_amdgcn_wmma_f32_16x16x32_f16(
                    false, a[t], false, bf, (short)0, acc[nt], false, false);
            }
        }
    }

    const int len = lengths[b];
    #pragma unroll
    for (int nt = 0; nt < 4; ++nt) {
        const int n  = nb + nt * 16 + Ml;
        const float bv = bias[n];
        #pragma unroll
        for (int r = 0; r < 8; ++r) {
            int s = s0 + r + 8 * hi;            // D: row M = r + 8*(lane>=16)
            float m = (s < len) ? 1.f : 0.f;
            float y = (acc[nt][r] + bv) * m;
            size_t o = ((size_t)b * S_ + s) * H_ + n;
            if (mode) out[o] = (y + hidden[o]) * RSQRT2_;
            else      out[o] = y;
        }
    }
}

// ---------------------------------------------------------------------------
extern "C" void kernel_launch(void* const* d_in, const int* in_sizes, int n_in,
                              void* d_out, int out_size, void* d_ws, size_t ws_size,
                              hipStream_t stream) {
    const float* hidden  = (const float*)d_in[0];
    const float* style   = (const float*)d_in[1];
    const int*   lengths = (const int*)  d_in[2];
    const float* p1w = (const float*)d_in[3];
    const float* p1b = (const float*)d_in[4];
    const float* p2w = (const float*)d_in[5];
    const float* p2b = (const float*)d_in[6];
    const float* c1v = (const float*)d_in[7];
    const float* c1g = (const float*)d_in[8];
    const float* c1b = (const float*)d_in[9];
    const float* c2v = (const float*)d_in[10];
    const float* c2g = (const float*)d_in[11];
    const float* c2b = (const float*)d_in[12];
    float* out = (float*)d_out;

    // workspace layout (256-aligned)
    char* ws = (char*)d_ws;
    _Float16* Wh   = (_Float16*)(ws);              // 2*3*512*512 f16  = 3 MB
    float*    scl  = (float*)(ws + 3145728);       // 1024 f32
    float*    hs   = (float*)(ws + 3149824);       // 2*8*1024 f32
    float*    mn1  = (float*)(ws + 3215360);       // 4096 f32
    float*    rs1  = mn1 + 4096;
    float*    mn2  = rs1 + 4096;
    float*    rs2  = mn2 + 4096;
    _Float16* hpad = (_Float16*)(ws + 3280896);    // 8*2050*512 f16 ≈ 16.8 MB

    // 1) weight-norm scales
    wn_scale_kernel<<<4, 256, 0, stream>>>(c1v, c1g, c2v, c2g, scl);
    // 2) pack f16 weights tap-major
    wn_write_kernel<<<6144, 256, 0, stream>>>(c1v, c2v, scl, Wh);
    // 3) style projections (both)
    proj_kernel<<<64, 256, 0, stream>>>(style, p1w, p1b, p2w, p2b, hs);
    // 4) AdaIN-1 stats on hidden_states
    stats_kernel<<<64, 256, 0, stream>>>(hidden, lengths, mn1, rs1);
    // 5) normalize+leaky -> padded f16
    norm_kernel<<<32800, 256, 0, stream>>>(hidden, hs, mn1, rs1, lengths, hpad);
    // 6) conv1 GEMM -> f32 in d_out (scratch): (y + b1) * mask
    {
        dim3 grid(512, 2, 1);
        gemm_conv3_kernel<<<grid, 256, 0, stream>>>(hpad, Wh, c1b, hidden,
                                                    lengths, out, 0);
    }
    // 7) AdaIN-2 stats on conv1 result
    stats_kernel<<<64, 256, 0, stream>>>(out, lengths, mn2, rs2);
    // 8) normalize+leaky -> padded f16 (reuse hpad), gamma/beta from proj2
    norm_kernel<<<32800, 256, 0, stream>>>(out, hs + 8 * 1024, mn2, rs2,
                                           lengths, hpad);
    // 9) conv2 GEMM + residual + 1/sqrt(2) -> d_out
    {
        dim3 grid(512, 2, 1);
        gemm_conv3_kernel<<<grid, 256, 0, stream>>>(hpad, Wh + 3 * 512 * 512,
                                                    c2b, hidden, lengths, out, 1);
    }
    (void)in_sizes; (void)n_in; (void)out_size; (void)ws_size;
}
